// GrainNN2_17549236371819
// MI455X (gfx1250) — matrix operand order repaired and checked by reading, not compiled, not run on hardware
//
#include <hip/hip_runtime.h>
#include <hip/hip_bf16.h>
#include <stdint.h>

// ---------------------------------------------------------------------------
// GrainNN2 hetero-graph LSTM for MI455X (gfx1250, wave32, WMMA).
//
// Strategy:
//  * Per cell & node type, fuse ALL gate GEMMs into one WMMA GEMM:
//      Z[N x 512] = A[N x K] @ Wcat[K x 512] + bias
//    where A = [x|agg_gj|agg_jj|h] (joint) or [x|agg_jg|h] (grain), f16,
//    Wcat pre-packed f16 with Wr_gj + Wr_jj combined (both multiply h),
//    gates stacked along N in sorted order (c,f,i,o). f32 accumulation.
//  * Edge aggregation (segment mean) via global_atomic_add_f32 scatter.
//  * GEMM: one wave32 per 32x64 tile, v_wmma_f32_16x16x32_f16, fragments
//    loaded per CDNA5 ISA VGPR layouts (A: K-octet per lane group; B from
//    N-major weights -> same K-octet split as A, two 8-half loads per lane).
// ---------------------------------------------------------------------------

#define HDIM 128

typedef _Float16 v16h __attribute__((ext_vector_type(16)));
typedef _Float16 v8h  __attribute__((ext_vector_type(8)));
typedef float    v8f  __attribute__((ext_vector_type(8)));

union FragH { v16h f; v8h h[2]; };

__device__ __forceinline__ float sigf(float x) { return 1.0f / (1.0f + __expf(-x)); }

// ----------------------------- utility kernels -----------------------------

__global__ void k_zero(uint32_t* __restrict__ p, long n) {
  long t = (long)blockIdx.x * blockDim.x + threadIdx.x;
  long s = (long)gridDim.x * blockDim.x;
  for (; t < n; t += s) p[t] = 0u;
}

__global__ void k_copy_f32(float* __restrict__ d, const float* __restrict__ s, long n) {
  long t = (long)blockIdx.x * blockDim.x + threadIdx.x;
  if (t < n) d[t] = s[t];
}

// ------------------------- edge scatter (segment sum) -----------------------
// one wave handles one edge: 32 lanes x float4 = 128 channels

__global__ void k_scatter(const float* __restrict__ h, const int* __restrict__ src,
                          const int* __restrict__ dst, int E,
                          float* __restrict__ sum, float* __restrict__ cnt) {
  long t = (long)blockIdx.x * blockDim.x + threadIdx.x;
  if (t >= (long)E * 32) return;
  int e = (int)(t >> 5), lane = (int)(t & 31);
  int s = src[e], d = dst[e];
  float4 hv = reinterpret_cast<const float4*>(h + (size_t)s * HDIM)[lane];
  float* o = sum + (size_t)d * HDIM + lane * 4;
  atomicAdd(o + 0, hv.x);
  atomicAdd(o + 1, hv.y);
  atomicAdd(o + 2, hv.z);
  atomicAdd(o + 3, hv.w);
  if (lane == 0) atomicAdd(cnt + d, 1.0f);
}

// ------------------------------ A-matrix packing ---------------------------

__global__ void k_copy_pack(const float* __restrict__ src, int sstr, int ncols,
                            _Float16* __restrict__ A, int lda, int coff, int N) {
  long t = (long)blockIdx.x * blockDim.x + threadIdx.x;
  long tot = (long)N * ncols;
  if (t >= tot) return;
  int n = (int)(t / ncols), c = (int)(t % ncols);
  A[(size_t)n * lda + coff + c] = (_Float16)src[(size_t)n * sstr + c];
}

__global__ void k_agg_pack(const float* __restrict__ sum, const float* __restrict__ cnt,
                           _Float16* __restrict__ A, int lda, int coff, int N) {
  long t = (long)blockIdx.x * blockDim.x + threadIdx.x;
  long tot = (long)N * HDIM;
  if (t >= tot) return;
  int n = (int)(t >> 7), c = (int)(t & 127);
  float cc = cnt[n];
  cc = cc > 1.0f ? cc : 1.0f;
  A[(size_t)n * lda + coff + c] = (_Float16)(sum[t] / cc);
}

// ------------------------------ weight packing ------------------------------
// Flat param layout (jax pytree, dicts alphabetical):
//  decoder[0],decoder[1],encoder[0],encoder[1],linear
//  layer -> gates c,f,i,o ; gate -> W_grain, W_joint, b_grain, b_joint,
//  conv{gj,jg,jj} each {Wl(128x128), Wr(128x128), bl(128)}  (Wl<Wr<bl sorted)

__global__ void k_pack_wj(const float* __restrict__ P, long gate0, long gstride,
                          int layer, _Float16* __restrict__ Wt, float* __restrict__ bias) {
  long t = (long)blockIdx.x * blockDim.x + threadIdx.x;
  if (t >= 512L * 512L) return;
  int n = (int)(t >> 9);  // output column (gate*128 + nc), N-major row of Wt
  int k = (int)(t & 511); // K index
  int g = n >> 7, nc = n & 127;
  long base = gate0 + (long)g * gstride;
  int ing = layer ? 128 : 8, inj = layer ? 128 : 6;
  long Wg = (long)ing * 128;          // W_grain size
  long Wjoff = Wg;                    // W_joint offset
  long conv = Wg + (long)inj * 128 + 256;
  long gjWl = conv, gjWr = conv + 16384, gjbl = conv + 32768;
  long jjWl = conv + 2 * 32896, jjWr = jjWl + 16384, jjbl = jjWl + 32768;
  int inc = layer ? 128 : 32;         // x slot width in A (l0 x padded to 32)
  float v = 0.0f;
  if (k < inc) {
    if (k < inj) v = P[base + Wjoff + (long)k * 128 + nc];
  } else if (k < inc + 128) {
    int kk = k - inc;            v = P[base + gjWl + (long)kk * 128 + nc];
  } else if (k < inc + 256) {
    int kk = k - inc - 128;      v = P[base + jjWl + (long)kk * 128 + nc];
  } else if (k < inc + 384) {
    int kk = k - inc - 256;
    v = P[base + gjWr + (long)kk * 128 + nc] + P[base + jjWr + (long)kk * 128 + nc];
  }
  Wt[(long)n * 512 + k] = (_Float16)v;
  if (k == 0) {
    long bj = Wg + (long)inj * 128 + 128;  // b_joint (after b_grain)
    bias[n] = P[base + bj + nc] + P[base + gjbl + nc] + P[base + jjbl + nc];
  }
}

__global__ void k_pack_wg(const float* __restrict__ P, long gate0, long gstride,
                          int layer, _Float16* __restrict__ Wt, float* __restrict__ bias) {
  long t = (long)blockIdx.x * blockDim.x + threadIdx.x;
  if (t >= 512L * 384L) return;
  int n = (int)(t / 384);
  int k = (int)(t % 384);
  int g = n >> 7, nc = n & 127;
  long base = gate0 + (long)g * gstride;
  int ing = layer ? 128 : 8, inj = layer ? 128 : 6;
  long conv = (long)(ing + inj) * 128 + 256;
  long jgWl = conv + 32896, jgWr = jgWl + 16384, jgbl = jgWl + 32768;
  int inc = layer ? 128 : 32;
  float v = 0.0f;
  if (k < inc) {
    if (k < ing) v = P[base + (long)k * 128 + nc];     // W_grain at gate base
  } else if (k < inc + 128) {
    int kk = k - inc;       v = P[base + jgWl + (long)kk * 128 + nc];
  } else if (k < inc + 256) {
    int kk = k - inc - 128; v = P[base + jgWr + (long)kk * 128 + nc];
  }
  Wt[(long)n * 384 + k] = (_Float16)v;
  if (k == 0)
    bias[n] = P[base + (long)(ing + inj) * 128 + nc] + P[base + jgbl + nc];
}

// --------------------------------- WMMA GEMM --------------------------------
// C[rows x 512] = A[rows x K](f16, lda) @ Wt^T  (Wt is N-major [512 x ldw] f16)
// One wave per 32x64 tile. grid = (8, rows/32), block = 32.

__global__ void k_gemm(const _Float16* __restrict__ A, int lda,
                       const _Float16* __restrict__ Wt, int ldw,
                       const float* __restrict__ bias,
                       float* __restrict__ C, int ldc, int K) {
  const int lane  = threadIdx.x & 31;
  const int mrow  = lane & 15;
  const int hi    = lane >> 4;
  const int koffA = hi * 8;    // A frag: lane group selects K octet (0 / 8)
  const int koffB = hi * 8;    // B frag: lane group selects K octet (0 / 8)
  const int r0 = blockIdx.y * 32;
  const int c0 = blockIdx.x * 64;

  v8f acc[2][4] = {};

  for (int k = 0; k < K; k += 32) {
    FragH a[2], b[4];
#pragma unroll
    for (int mt = 0; mt < 2; ++mt) {
      const _Float16* p = A + (size_t)(r0 + mt * 16 + mrow) * lda + k + koffA;
      a[mt].h[0] = *reinterpret_cast<const v8h*>(p);       // K+0..7 (or 8..15)
      a[mt].h[1] = *reinterpret_cast<const v8h*>(p + 16);  // K+16..23 (24..31)
    }
#pragma unroll
    for (int nt = 0; nt < 4; ++nt) {
      const _Float16* p = Wt + (size_t)(c0 + nt * 16 + mrow) * ldw + k + koffB;
      b[nt].h[0] = *reinterpret_cast<const v8h*>(p);       // K+0..7 (or 8..15)
      b[nt].h[1] = *reinterpret_cast<const v8h*>(p + 16);  // K+16..23 (24..31)
    }
#pragma unroll
    for (int mt = 0; mt < 2; ++mt)
#pragma unroll
      for (int nt = 0; nt < 4; ++nt)
        acc[mt][nt] = __builtin_amdgcn_wmma_f32_16x16x32_f16(
            false, a[mt].f, false, b[nt].f, (short)0, acc[mt][nt], false, false);
  }

#pragma unroll
  for (int nt = 0; nt < 4; ++nt) {
    int col = c0 + nt * 16 + mrow;
    float bv = bias[col];
#pragma unroll
    for (int mt = 0; mt < 2; ++mt) {
#pragma unroll
      for (int v = 0; v < 8; ++v) {
        int row = r0 + mt * 16 + v + hi * 8;   // C layout: VGPR v -> M=v / v+8
        C[(size_t)row * ldc + col] = acc[mt][nt][v] + bv;
      }
    }
  }
}

// ------------------------------- LSTM pointwise ----------------------------
// Z cols: [0:128)=c gate, [128:256)=f, [256:384)=i, [384:512)=o (sorted order)

__global__ void k_lstm(const float* __restrict__ Z, float* __restrict__ h,
                       float* __restrict__ c, int N) {
  long t = (long)blockIdx.x * blockDim.x + threadIdx.x;
  if (t >= (long)N * HDIM) return;
  int n = (int)(t >> 7), ch = (int)(t & 127);
  const float* z = Z + (size_t)n * 512;
  float zc = z[ch], zf = z[128 + ch], zi = z[256 + ch], zo = z[384 + ch];
  float cn = sigf(zf) * c[t] + sigf(zi) * tanhf(zc);
  h[t] = sigf(zo) * tanhf(cn);
  c[t] = cn;
}

// -------------------------------- output head ------------------------------

__global__ void k_linear(const float* __restrict__ h, const float* __restrict__ P,
                         long offW, long offB, float* __restrict__ y, int N) {
  int n = blockIdx.x * blockDim.x + threadIdx.x;
  if (n >= N) return;
  const float* hp = h + (size_t)n * HDIM;
  float y0 = P[offB], y1 = P[offB + 1];
  for (int k = 0; k < HDIM; ++k) {
    float hv = hp[k];
    y0 += hv * P[offW + 2 * k];
    y1 += hv * P[offW + 2 * k + 1];
  }
  y[2 * n] = y0;
  y[2 * n + 1] = y1;
}

__global__ void k_joint_out(const float* __restrict__ yr, float* __restrict__ xj,
                            float* __restrict__ out, int N) {
  int n = blockIdx.x * blockDim.x + threadIdx.x;
  if (n >= N) return;
  float s0 = sigf(yr[2 * n]) - 0.5f;
  float s1 = sigf(yr[2 * n + 1]) - 0.5f;
  out[2 * n] = s0;
  out[2 * n + 1] = s1;
  float* x = xj + (size_t)n * 6;
  x[0] += s0; x[1] += s1; x[4] = s0; x[5] = s1;
}

__global__ void k_grain_area(const float* __restrict__ yr, const float* __restrict__ xg,
                             float* __restrict__ area, float* __restrict__ norm, int N) {
  int n = blockIdx.x * blockDim.x + threadIdx.x;
  if (n >= N) return;
  float a = yr[2 * n] + xg[(size_t)n * 8 + 3];
  a = a > 0.0f ? a : 0.0f;   // relu; |a| == a
  area[n] = a;
  atomicAdd(norm, a);
}

__global__ void k_grain_fin(const float* __restrict__ yr, const float* __restrict__ area,
                            const float* __restrict__ norm, float* __restrict__ xg,
                            float* __restrict__ out, int N) {
  int n = blockIdx.x * blockDim.x + threadIdx.x;
  if (n >= N) return;
  float denom = fmaxf(norm[0], 1e-12f);
  float x3 = xg[(size_t)n * 8 + 3];
  float yg0 = area[n] / denom - x3;
  float t = yr[2 * n + 1];
  float yg1 = t > 0.0f ? t : 0.0f;
  out[2 * n] = yg0;
  out[2 * n + 1] = yg1;
  xg[(size_t)n * 8 + 3] = x3 + yg0;
  xg[(size_t)n * 8 + 4] = yg1;
  xg[(size_t)n * 8 + 7] = yg0;
}

// --------------------------------- host side --------------------------------

static inline int gblk(long n, int bs) { return (int)((n + bs - 1) / bs); }

extern "C" void kernel_launch(void* const* d_in, const int* in_sizes, int n_in,
                              void* d_out, int out_size, void* d_ws, size_t ws_size,
                              hipStream_t stream) {
  const float* xg_in = (const float*)d_in[0];
  const float* xj_in = (const float*)d_in[1];
  const float* P     = (const float*)d_in[2];
  const int* egj = (const int*)d_in[3];
  const int* ejg = (const int*)d_in[4];
  const int* ejj = (const int*)d_in[5];

  const int Ng = in_sizes[0] / 8;
  const int Nj = in_sizes[1] / 6;
  const int E  = in_sizes[3] / 2;
  const int Ngp = (Ng + 31) & ~31;
  const int Njp = (Nj + 31) & ~31;

  // param layout constants
  const long gsz0 = (8 + 6) * 128 + 256 + 3 * 32896;        // 100736
  const long gsz1 = (128 + 128) * 128 + 256 + 3 * 32896;    // 131712
  const long stackSz = 4 * (gsz0 + gsz1);                   // 929792
  const long linBase = 2 * stackSz;                         // linear params

  // bump allocator on workspace (~800 MB total)
  char* wp = (char*)d_ws;
  auto alloc = [&](size_t bytes) -> char* {
    char* r = wp;
    wp += (bytes + 255) & ~(size_t)255;
    return r;
  };
  float* xg = (float*)alloc(sizeof(float) * (size_t)Ng * 8);
  float* xj = (float*)alloc(sizeof(float) * (size_t)Nj * 6);
  float* hc = (float*)alloc(sizeof(float) * ((size_t)Ng * 128 * 4 + (size_t)Nj * 128 * 4));
  float* h0g = hc;
  float* c0g = h0g + (size_t)Ng * 128;
  float* h1g = c0g + (size_t)Ng * 128;
  float* c1g = h1g + (size_t)Ng * 128;
  float* h0j = c1g + (size_t)Ng * 128;
  float* c0j = h0j + (size_t)Nj * 128;
  float* h1j = c0j + (size_t)Nj * 128;
  float* c1j = h1j + (size_t)Nj * 128;
  const long aggWords = (long)Nj * 128 * 2 + (long)Ng * 128 + 2L * Nj + Ng;
  float* agg = (float*)alloc(sizeof(float) * (size_t)aggWords);
  float* aggGJ = agg;
  float* aggJJ = aggGJ + (size_t)Nj * 128;
  float* aggJG = aggJJ + (size_t)Nj * 128;
  float* cntGJ = aggJG + (size_t)Ng * 128;
  float* cntJJ = cntGJ + Nj;
  float* cntJG = cntJJ + Nj;
  _Float16* Aj = (_Float16*)alloc(2 * (size_t)Njp * 512);
  _Float16* Ag = (_Float16*)alloc(2 * (size_t)Ngp * 384);
  float* Z = (float*)alloc(4 * (size_t)Njp * 512);          // shared joint/grain
  _Float16* wtj[4]; _Float16* wtg[4]; float* bj[4]; float* bg[4];
  for (int i = 0; i < 4; ++i) {
    wtj[i] = (_Float16*)alloc(2 * 512 * 512);
    wtg[i] = (_Float16*)alloc(2 * 512 * 384);
    bj[i] = (float*)alloc(4 * 512);
    bg[i] = (float*)alloc(4 * 512);
  }
  float* yrg  = (float*)alloc(4 * (size_t)Ng * 2);
  float* yrj  = (float*)alloc(4 * (size_t)Nj * 2);
  float* area = (float*)alloc(4 * (size_t)Ng);
  float* norm = (float*)alloc(256);

  float* outG = (float*)d_out;
  float* outJ = outG + (size_t)Ng * 2;

  auto zero = [&](void* p, long words) {
    long b = (words + 255) / 256;
    if (b > 32768) b = 32768;
    k_zero<<<dim3((uint32_t)b), dim3(256), 0, stream>>>((uint32_t*)p, words);
  };

  // ---- pack weights once per (stack, layer): idx = stack*2 + layer; stack0=decoder
  for (int s = 0; s < 2; ++s)
    for (int l = 0; l < 2; ++l) {
      int idx = s * 2 + l;
      long gate0 = (long)s * stackSz + (l ? 4 * gsz0 : 0);
      long gstr = l ? gsz1 : gsz0;
      k_pack_wj<<<gblk(512L * 512, 256), 256, 0, stream>>>(P, gate0, gstr, l, wtj[idx], bj[idx]);
      k_pack_wg<<<gblk(512L * 384, 256), 256, 0, stream>>>(P, gate0, gstr, l, wtg[idx], bg[idx]);
    }

  // ---- init mutable x and hidden state
  k_copy_f32<<<gblk((long)Ng * 8, 256), 256, 0, stream>>>(xg, xg_in, (long)Ng * 8);
  k_copy_f32<<<gblk((long)Nj * 6, 256), 256, 0, stream>>>(xj, xj_in, (long)Nj * 6);
  zero(hc, ((long)Ng + Nj) * 128 * 4);

  // ---- one LSTM graph-cell
  auto cell = [&](int s, int l,
                  const float* curG, int strG, int colsG,
                  const float* curJ, int strJ, int colsJ,
                  float* hg, float* cg, float* hj, float* cj) {
    int idx = s * 2 + l;
    int inc = l ? 128 : 32;
    int Kj = inc + 384, Kg = inc + 256;
    zero(agg, aggWords);
    zero(Aj, (long)Njp * 256);   // halves -> dwords
    zero(Ag, (long)Ngp * 192);
    long st = (long)E * 32;
    k_scatter<<<gblk(st, 256), 256, 0, stream>>>(hg, egj, egj + E, E, aggGJ, cntGJ);
    k_scatter<<<gblk(st, 256), 256, 0, stream>>>(hj, ejg, ejg + E, E, aggJG, cntJG);
    k_scatter<<<gblk(st, 256), 256, 0, stream>>>(hj, ejj, ejj + E, E, aggJJ, cntJJ);
    // pack A_joint = [cur | agg_gj | agg_jj | h]
    k_copy_pack<<<gblk((long)Nj * colsJ, 256), 256, 0, stream>>>(curJ, strJ, colsJ, Aj, 512, 0, Nj);
    k_agg_pack<<<gblk((long)Nj * 128, 256), 256, 0, stream>>>(aggGJ, cntGJ, Aj, 512, inc, Nj);
    k_agg_pack<<<gblk((long)Nj * 128, 256), 256, 0, stream>>>(aggJJ, cntJJ, Aj, 512, inc + 128, Nj);
    k_copy_pack<<<gblk((long)Nj * 128, 256), 256, 0, stream>>>(hj, 128, 128, Aj, 512, inc + 256, Nj);
    // pack A_grain = [cur | agg_jg | h]
    k_copy_pack<<<gblk((long)Ng * colsG, 256), 256, 0, stream>>>(curG, strG, colsG, Ag, 384, 0, Ng);
    k_agg_pack<<<gblk((long)Ng * 128, 256), 256, 0, stream>>>(aggJG, cntJG, Ag, 384, inc, Ng);
    k_copy_pack<<<gblk((long)Ng * 128, 256), 256, 0, stream>>>(hg, 128, 128, Ag, 384, inc + 128, Ng);
    // fused gate GEMMs (WMMA) + pointwise LSTM; Z reused joint then grain
    dim3 gj(8, Njp / 32);
    k_gemm<<<gj, 32, 0, stream>>>(Aj, 512, wtj[idx], 512, bj[idx], Z, 512, Kj);
    k_lstm<<<gblk((long)Nj * 128, 256), 256, 0, stream>>>(Z, hj, cj, Nj);
    dim3 gg(8, Ngp / 32);
    k_gemm<<<gg, 32, 0, stream>>>(Ag, 384, wtg[idx], 384, bg[idx], Z, 512, Kg);
    k_lstm<<<gblk((long)Ng * 128, 256), 256, 0, stream>>>(Z, hg, cg, Ng);
  };

  // ---- encoder (stack index 1 in flat layout: 'decoder' < 'encoder')
  cell(1, 0, xg, 8, 8, xj, 6, 6, h0g, c0g, h0j, c0j);
  cell(1, 1, h0g, 128, 128, h0j, 128, 128, h1g, c1g, h1j, c1j);

  // ---- decoder, OUT_WIN = 2 steps
  for (int step = 0; step < 2; ++step) {
    cell(0, 0, xg, 8, 8, xj, 6, 6, h0g, c0g, h0j, c0j);
    cell(0, 1, h0g, 128, 128, h0j, 128, 128, h1g, c1g, h1j, c1j);
    // linear heads (grain W@linBase, b@+256; joint W@+258, b@+258+256)
    k_linear<<<gblk(Ng, 256), 256, 0, stream>>>(h1g, P, linBase, linBase + 256, yrg, Ng);
    k_linear<<<gblk(Nj, 256), 256, 0, stream>>>(h1j, P, linBase + 258, linBase + 258 + 256, yrj, Nj);
    k_joint_out<<<gblk(Nj, 256), 256, 0, stream>>>(yrj, xj, outJ, Nj);
    zero(norm, 1);
    k_grain_area<<<gblk(Ng, 256), 256, 0, stream>>>(yrg, xg, area, norm, Ng);
    k_grain_fin<<<gblk(Ng, 256), 256, 0, stream>>>(yrg, area, norm, xg, outG, Ng);
  }
  (void)n_in; (void)out_size; (void)ws_size;
}
